// VSSM_1924145348928
// MI455X (gfx1250) — matrix operand order repaired and checked by reading, not compiled, not run on hardware
//
#include <hip/hip_runtime.h>
#include <hip/hip_bf16.h>
#include <math.h>

// ---------------- problem constants (from reference) ----------------
constexpr int CB  = 2;             // batch
constexpr int CH  = 48;            // H
constexpr int CW  = 48;            // W
constexpr int CDM = 192;           // d_model
constexpr int CN  = 16;            // d_state
constexpr int CDI = 384;           // d_inner
constexpr int CR  = 12;            // dt_rank
constexpr int CK  = 4;             // scan directions
constexpr int CL  = CH * CW;       // 2304
constexpr int CBL = CB * CL;       // 4608
constexpr int CPROJ = CR + 2 * CN; // 44 rows of x_dbl
constexpr int CRP = 32;            // dt_rank padded to K-multiple of 32
constexpr int CMP = 48;            // 44 padded to 48

typedef __attribute__((ext_vector_type(16))) _Float16 v16h;
typedef __attribute__((ext_vector_type(8)))  _Float16 v8h;
typedef __attribute__((ext_vector_type(8)))  float    v8f;

// ---------------------------------------------------------------
// Generic pack: dst (f16, Mp x Kp row-major, zero-padded) from strided
// f32 src.  Also used for B^T packing (swap roles of strides).
// Batched over gridDim.y with z = kd*nb + b decomposition.
// ---------------------------------------------------------------
__global__ void pack_f16_kernel(const float* __restrict__ src,
                                _Float16* __restrict__ dst,
                                int M, int K, int Mp, int Kp,
                                long sm, long sk,
                                long bsk, long bsb, long bdst, int nb)
{
    const int z  = blockIdx.y;
    const int kd = z / nb, bb = z % nb;
    src += (long)kd * bsk + (long)bb * bsb;
    dst += (long)z * bdst;

    long id = (long)blockIdx.x * blockDim.x + threadIdx.x;
    const long tot = (long)Mp * Kp;
    if (id >= tot) return;
    const int k = (int)(id % Kp);
    const int m = (int)(id / Kp);
    float v = (m < M && k < K) ? src[(long)m * sm + (long)k * sk] : 0.0f;
    dst[id] = (_Float16)v;
}

// ---------------------------------------------------------------
// Packed WMMA GEMM: C[m,n] = sum_k A[m,k] * B[k,n]
//   Ap : f16, Mp x Kp row-major (zero padded)
//   Btp: f16, N  x Kp row-major (= B^T, zero padded in K)
// One wave per 16x64 C tile: A fragment reused across 4 v_wmma.
// Branch-free inner loop: 2x b128 (A) + 4x 32B (B) loads per chunk.
// ---------------------------------------------------------------
__global__ __launch_bounds__(32)
void gemm_wmma_packed(const _Float16* __restrict__ Ap,
                      const _Float16* __restrict__ Btp,
                      float* __restrict__ C,
                      int M, int Kp, long ldc_m, long ldc_n,
                      long bAk, long bAb, long bBk, long bBb,
                      long bCk, long bCb, int nb)
{
    const int z  = blockIdx.z;
    const int kd = z / nb, bb = z % nb;
    Ap  += (long)kd * bAk + (long)bb * bAb;
    Btp += (long)kd * bBk + (long)bb * bBb;
    C   += (long)kd * bCk + (long)bb * bCb;

    const int lane = threadIdx.x;          // wave32
    const int l16  = lane & 15;
    const int hi   = lane >> 4;
    const int m0   = blockIdx.y * 16;
    const int n0   = blockIdx.x * 64;

    // A 16x32 f16 fragment: row m, k = k0 + laneHalf*16 + 0..15 (contiguous)
    const _Float16* arow = Ap  + (long)(m0 + l16) * Kp + hi * 16;
    // B 32x16 fragment via B^T rows: col n, k = k0 + laneHalf*16 + 0..15 (contiguous)
    const _Float16* brow = Btp + (long)(n0 + l16) * Kp + hi * 16;
    const long bstep = (long)16 * Kp;

    v8f acc0 = {}, acc1 = {}, acc2 = {}, acc3 = {};
    for (int k0 = 0; k0 < Kp; k0 += 32) {
        const v8h alo = *(const v8h*)(arow + k0);
        const v8h ahi = *(const v8h*)(arow + k0 + 8);
        const v16h af = __builtin_shufflevector(alo, ahi,
                            0,1,2,3,4,5,6,7,8,9,10,11,12,13,14,15);
        const v16h b0 = *(const v16h*)(brow + k0);
        const v16h b1 = *(const v16h*)(brow + bstep + k0);
        const v16h b2 = *(const v16h*)(brow + 2 * bstep + k0);
        const v16h b3 = *(const v16h*)(brow + 3 * bstep + k0);
        acc0 = __builtin_amdgcn_wmma_f32_16x16x32_f16(false, af, false, b0, (short)0, acc0, false, false);
        acc1 = __builtin_amdgcn_wmma_f32_16x16x32_f16(false, af, false, b1, (short)0, acc1, false, false);
        acc2 = __builtin_amdgcn_wmma_f32_16x16x32_f16(false, af, false, b2, (short)0, acc2, false, false);
        acc3 = __builtin_amdgcn_wmma_f32_16x16x32_f16(false, af, false, b3, (short)0, acc3, false, false);
    }

    const int ncol = n0 + l16;
#pragma unroll
    for (int r = 0; r < 8; ++r) {
        const int m = m0 + r + (hi << 3);  // C layout: M split 8/8 per lane half
        if (m < M) {
            float* crow = C + (long)m * ldc_m;
            crow[(long)(ncol)      * ldc_n] = acc0[r];
            crow[(long)(ncol + 16) * ldc_n] = acc1[r];
            crow[(long)(ncol + 32) * ldc_n] = acc2[r];
            crow[(long)(ncol + 48) * ldc_n] = acc3[r];
        }
    }
}

// ---------------------------------------------------------------
// Depthwise 3x3 conv + bias + SiLU.  Input xz (B,L,2*DI) channels 0..DI-1,
// output xc (B,DI,L).
// ---------------------------------------------------------------
__global__ void conv_silu_kernel(const float* __restrict__ xz,
                                 const float* __restrict__ cw,
                                 const float* __restrict__ cb,
                                 float* __restrict__ xc)
{
    long id = (long)blockIdx.x * blockDim.x + threadIdx.x;
    const long tot = (long)CB * CDI * CL;
    if (id >= tot) return;
    const int l = (int)(id % CL);
    long t = id / CL;
    const int d = (int)(t % CDI);
    const int b = (int)(t / CDI);
    const int h = l / CW, w = l % CW;

    float s = cb[d];
#pragma unroll
    for (int dh = -1; dh <= 1; ++dh) {
#pragma unroll
        for (int dw = -1; dw <= 1; ++dw) {
            const int hh = h + dh, ww = w + dw;
            if (hh >= 0 && hh < CH && ww >= 0 && ww < CW) {
                const float xv = xz[((long)b * CL + hh * CW + ww) * (2 * CDI) + d];
                s += xv * cw[d * 9 + (dh + 1) * 3 + (dw + 1)];
            }
        }
    }
    s = s / (1.0f + expf(-s));   // SiLU
    xc[id] = s;
}

// ---------------------------------------------------------------
// Cross-scan: build xs (B,4,DI,L) from xc (B,DI,L) via index remaps.
// ---------------------------------------------------------------
__global__ void cross_scan_kernel(const float* __restrict__ xc,
                                  float* __restrict__ xs)
{
    long id = (long)blockIdx.x * blockDim.x + threadIdx.x;
    const long tot = (long)CB * CK * CDI * CL;
    if (id >= tot) return;
    const int l = (int)(id % CL);
    long t = id / CL;
    const int d = (int)(t % CDI); t /= CDI;
    const int k = (int)(t % CK);
    const int b = (int)(t / CK);

    int src;
    if (k == 0) {
        src = l;
    } else if (k == 1) {                      // transposed raster
        const int w = l / CH, h = l % CH;
        src = h * CW + w;
    } else if (k == 2) {                      // reversed raster
        src = CL - 1 - l;
    } else {                                  // reversed transposed raster
        const int l2 = CL - 1 - l;
        const int w = l2 / CH, h = l2 % CH;
        src = h * CW + w;
    }
    xs[id] = xc[((long)b * CDI + d) * CL + src];
}

// ---------------------------------------------------------------
// Selective scan: one thread per (b,k,d) sequence, 16 hidden states in
// VGPRs; shared B/C selection matrices staged in LDS per 64-step chunk
// (each 128-thread block maps to a single (b,k) since DI = 3*128).
// ---------------------------------------------------------------
#define SCAN_CHUNK 64
__global__ __launch_bounds__(128)
void scan_kernel(const float* __restrict__ xs,
                 const float* __restrict__ dts,
                 const float* __restrict__ xdbl,
                 const float* __restrict__ A_logs,
                 const float* __restrict__ Ds,
                 const float* __restrict__ dt_bias,
                 float* __restrict__ outy)
{
    __shared__ float sB[CN * SCAN_CHUNK];
    __shared__ float sC[CN * SCAN_CHUNK];

    const int tid = threadIdx.x;
    const long s = (long)blockIdx.x * blockDim.x + tid;   // < CB*CK*CDI = 3072
    const int d = (int)(s % CDI);
    long t = s / CDI;
    const int k = (int)(t % CK);
    const int b = (int)(t / CK);

    const float* u   = xs   + ((long)(b * CK + k) * CDI + d) * CL;
    const float* dtp = dts  + ((long)(b * CK + k) * CDI + d) * CL;
    const float* Bsr = xdbl + ((long)(b * CK + k) * CPROJ + CR) * CL;       // rows 12..27
    const float* Csr = xdbl + ((long)(b * CK + k) * CPROJ + CR + CN) * CL;  // rows 28..43
    float*       yo  = outy + ((long)(b * CK + k) * CDI + d) * CL;

    float Arow[CN];
#pragma unroll
    for (int n = 0; n < CN; ++n)
        Arow[n] = -expf(A_logs[((long)k * CDI + d) * CN + n]);
    const float Dv   = Ds[k * CDI + d];
    const float bias = dt_bias[k * CDI + d];

    float h[CN];
#pragma unroll
    for (int n = 0; n < CN; ++n) h[n] = 0.0f;

    for (int l0 = 0; l0 < CL; l0 += SCAN_CHUNK) {
        if (l0 + SCAN_CHUNK < CL) {            // prefetch next chunk of u / dt
            __builtin_prefetch(u + l0 + SCAN_CHUNK, 0, 1);
            __builtin_prefetch(dtp + l0 + SCAN_CHUNK, 0, 1);
        }
        __syncthreads();
        for (int i = tid; i < CN * SCAN_CHUNK; i += blockDim.x) {
            const int n = i / SCAN_CHUNK, j = i % SCAN_CHUNK;
            sB[i] = Bsr[(long)n * CL + l0 + j];
            sC[i] = Csr[(long)n * CL + l0 + j];
        }
        __syncthreads();

        for (int j = 0; j < SCAN_CHUNK; ++j) {
            const int l = l0 + j;
            const float x   = dtp[l] + bias;
            const float dlt = (x > 20.0f) ? x : log1pf(expf(x));  // softplus
            const float ul  = u[l];
            const float du  = dlt * ul;
            float y = 0.0f;
#pragma unroll
            for (int n = 0; n < CN; ++n) {
                const float dA = expf(dlt * Arow[n]);
                h[n] = dA * h[n] + du * sB[n * SCAN_CHUNK + j];
                y   += h[n] * sC[n * SCAN_CHUNK + j];
            }
            yo[l] = y + Dv * ul;
        }
    }
}

// ---------------------------------------------------------------
// Merge 4 directions + LayerNorm over DI + SiLU gate.  One block per
// (b,l) row, 384 threads, LDS tree reduction for mean/var.
// Writes yin16 (BL x DI, f16 row-major) -> packed A for out_proj GEMM.
// ---------------------------------------------------------------
__global__ __launch_bounds__(CDI)
void merge_ln_gate_kernel(const float* __restrict__ outy,
                          const float* __restrict__ xz,
                          const float* __restrict__ g,
                          const float* __restrict__ bt,
                          _Float16* __restrict__ yin16)
{
    __shared__ float s1[CDI];
    __shared__ float s2[CDI];

    const int bl = blockIdx.x;          // 0..CBL-1
    const int b = bl / CL, l = bl % CL;
    const int d = threadIdx.x;          // 0..383
    const int h = l / CW, w = l % CW;
    const int lt = w * CH + h;          // transposed raster position

    const long base = (long)b * CK * CDI * CL;
    float v = outy[base + ((long)0 * CDI + d) * CL + l]
            + outy[base + ((long)2 * CDI + d) * CL + (CL - 1 - l)]
            + outy[base + ((long)1 * CDI + d) * CL + lt]
            + outy[base + ((long)3 * CDI + d) * CL + (CL - 1 - lt)];

    s1[d] = v;
    s2[d] = v * v;
    __syncthreads();
    if (d < 128) {                      // 384 -> 128
        s1[d] += s1[d + 128] + s1[d + 256];
        s2[d] += s2[d + 128] + s2[d + 256];
    }
    __syncthreads();
    for (int st = 64; st > 0; st >>= 1) {
        if (d < st) { s1[d] += s1[d + st]; s2[d] += s2[d + st]; }
        __syncthreads();
    }
    const float mu  = s1[0] * (1.0f / CDI);
    const float var = s2[0] * (1.0f / CDI) - mu * mu;

    const float yn = (v - mu) * rsqrtf(var + 1e-5f) * g[d] + bt[d];
    const float zv = xz[((long)b * CL + l) * (2 * CDI) + CDI + d];
    const float sig = 1.0f / (1.0f + expf(-zv));
    yin16[((long)b * CL + l) * CDI + d] = (_Float16)(yn * (zv * sig));
}

// ---------------------------------------------------------------
extern "C" void kernel_launch(void* const* d_in, const int* in_sizes, int n_in,
                              void* d_out, int out_size, void* d_ws, size_t ws_size,
                              hipStream_t stream)
{
    const float* x          = (const float*)d_in[0];   // (B,H,W,DM)
    const float* in_proj_w  = (const float*)d_in[1];   // (768,192)
    const float* conv_w     = (const float*)d_in[2];   // (384,1,3,3)
    const float* conv_b     = (const float*)d_in[3];   // (384)
    const float* x_proj_w   = (const float*)d_in[4];   // (4,44,384)
    const float* dt_projs_w = (const float*)d_in[5];   // (4,384,12)
    const float* dt_projs_b = (const float*)d_in[6];   // (4,384)
    const float* A_logs     = (const float*)d_in[7];   // (1536,16)
    const float* Ds         = (const float*)d_in[8];   // (1536)
    const float* out_norm_g = (const float*)d_in[9];   // (384)
    const float* out_norm_b = (const float*)d_in[10];  // (384)
    const float* out_proj_w = (const float*)d_in[11];  // (192,384)
    float* out = (float*)d_out;

    // ---------------- workspace carve-up ----------------
    float* ws   = (float*)d_ws;
    float* xz   = ws;                                     // BL * 768
    float* xc   = xz   + (long)CBL * (2 * CDI);           // B*DI*L
    float* xs   = xc   + (long)CB * CDI * CL;             // B*4*DI*L
    float* xdbl = xs   + (long)CB * CK * CDI * CL;        // B*4*44*L
    float* dts  = xdbl + (long)CB * CK * CPROJ * CL;      // B*4*DI*L
    float* outy = dts  + (long)CB * CK * CDI * CL;        // B*4*DI*L

    _Float16* f16base = (_Float16*)(outy + (long)CB * CK * CDI * CL);
    _Float16* Axp       = f16base;                              // 4608 x 192
    _Float16* Bt_inproj = Axp       + (long)CBL * CDM;          // 768  x 192
    _Float16* A_xdbl    = Bt_inproj + (long)(2*CDI) * CDM;      // 4 x 48 x 384
    _Float16* Bt_xdbl   = A_xdbl    + (long)CK * CMP * CDI;     // 8 x 2304 x 384
    _Float16* A_dt      = Bt_xdbl   + (long)CK*CB * CL * CDI;   // 4 x 384 x 32
    _Float16* Bt_dt     = A_dt      + (long)CK * CDI * CRP;     // 8 x 2304 x 32
    _Float16* Bt_out    = Bt_dt     + (long)CK*CB * CL * CRP;   // 192 x 384
    _Float16* yin16     = Bt_out    + (long)CDM * CDI;          // 4608 x 384

    // ---------------- 0) pack static weights + x ----------------
    {   // x -> Axp (4608 x 192)
        long tot = (long)CBL * CDM;
        pack_f16_kernel<<<dim3((tot + 255) / 256, 1), 256, 0, stream>>>(
            x, Axp, CBL, CDM, CBL, CDM, CDM, 1, 0, 0, 0, 1);
    }
    {   // in_proj_w -> Bt_inproj (768 x 192)  (already row-major (n,k))
        long tot = (long)(2 * CDI) * CDM;
        pack_f16_kernel<<<dim3((tot + 255) / 256, 1), 256, 0, stream>>>(
            in_proj_w, Bt_inproj, 2 * CDI, CDM, 2 * CDI, CDM, CDM, 1, 0, 0, 0, 1);
    }
    {   // x_proj_w -> A_xdbl (4 batches of 48 x 384)
        long tot = (long)CMP * CDI;
        pack_f16_kernel<<<dim3((tot + 255) / 256, CK), 256, 0, stream>>>(
            x_proj_w, A_xdbl, CPROJ, CDI, CMP, CDI, CDI, 1,
            (long)CPROJ * CDI, 0, (long)CMP * CDI, 1);
    }
    {   // dt_projs_w -> A_dt (4 batches of 384 x 32)
        long tot = (long)CDI * CRP;
        pack_f16_kernel<<<dim3((tot + 255) / 256, CK), 256, 0, stream>>>(
            dt_projs_w, A_dt, CDI, CR, CDI, CRP, CR, 1,
            (long)CDI * CR, 0, (long)CDI * CRP, 1);
    }
    {   // out_proj_w -> Bt_out (192 x 384)  (already row-major (n,k))
        long tot = (long)CDM * CDI;
        pack_f16_kernel<<<dim3((tot + 255) / 256, 1), 256, 0, stream>>>(
            out_proj_w, Bt_out, CDM, CDI, CDM, CDI, CDI, 1, 0, 0, 0, 1);
    }

    // ---------------- 1) in_proj: xz[BL,768] = x @ W^T ----------------
    {
        dim3 grid((2 * CDI) / 64, CBL / 16, 1);
        gemm_wmma_packed<<<grid, 32, 0, stream>>>(
            Axp, Bt_inproj, xz,
            CBL, CDM, 2 * CDI, 1,
            0, 0, 0, 0, 0, 0, 1);
    }

    // ---------------- 2) depthwise conv + SiLU -> xc ----------------
    {
        const long tot = (long)CB * CDI * CL;
        conv_silu_kernel<<<dim3((tot + 255) / 256), 256, 0, stream>>>(xz, conv_w, conv_b, xc);
    }

    // ---------------- 3) cross-scan -> xs ----------------
    {
        const long tot = (long)CB * CK * CDI * CL;
        cross_scan_kernel<<<dim3((tot + 255) / 256), 256, 0, stream>>>(xc, xs);
    }

    // ---------------- 4a) pack xs^T -> Bt_xdbl (8 x 2304 x 384) ----------------
    {
        long tot = (long)CL * CDI;
        pack_f16_kernel<<<dim3((tot + 255) / 256, CK * CB), 256, 0, stream>>>(
            xs, Bt_xdbl, CL, CDI, CL, CDI, 1, CL,
            (long)CDI * CL, (long)CK * CDI * CL, (long)CL * CDI, CB);
    }

    // ---------------- 4b) x_dbl[b,k] (44,L) = x_proj_w[k] @ xs[b,k] ----------------
    {
        dim3 grid(CL / 64, CMP / 16, CK * CB);
        gemm_wmma_packed<<<grid, 32, 0, stream>>>(
            A_xdbl, Bt_xdbl, xdbl,
            CPROJ, CDI, CL, 1,
            (long)CMP * CDI, 0,                              // A per-direction
            (long)CB * CL * CDI, (long)CL * CDI,             // B per z = kd*2+b
            (long)CPROJ * CL, (long)CK * CPROJ * CL,         // C
            CB);
    }

    // ---------------- 5a) pack x_dbl[0:12]^T -> Bt_dt (8 x 2304 x 32) ----------------
    {
        long tot = (long)CL * CRP;
        pack_f16_kernel<<<dim3((tot + 255) / 256, CK * CB), 256, 0, stream>>>(
            xdbl, Bt_dt, CL, CR, CL, CRP, 1, CL,
            (long)CPROJ * CL, (long)CK * CPROJ * CL, (long)CL * CRP, CB);
    }

    // ---------------- 5b) dts[b,k] (384,L) = dt_projs_w[k] @ x_dbl[b,k][0:12] ----------------
    {
        dim3 grid(CL / 64, CDI / 16, CK * CB);
        gemm_wmma_packed<<<grid, 32, 0, stream>>>(
            A_dt, Bt_dt, dts,
            CDI, CRP, CL, 1,
            (long)CDI * CRP, 0,
            (long)CB * CL * CRP, (long)CL * CRP,
            (long)CDI * CL, (long)CK * CDI * CL,
            CB);
    }

    // ---------------- 6) selective scan -> outy ----------------
    {
        const int tot = CB * CK * CDI;                  // 3072 threads
        scan_kernel<<<dim3(tot / 128), 128, 0, stream>>>(
            xs, dts, xdbl, A_logs, Ds, dt_projs_b, outy);
    }

    // ---------------- 7) merge + LayerNorm + gate -> yin16 (f16) ----------------
    {
        merge_ln_gate_kernel<<<dim3(CBL), CDI, 0, stream>>>(
            outy, xz, out_norm_g, out_norm_b, yin16);
    }

    // ---------------- 8) out_proj: out[BL,192] = yin @ W^T ----------------
    {
        dim3 grid(CDM / 64, CBL / 16, 1);
        gemm_wmma_packed<<<grid, 32, 0, stream>>>(
            yin16, Bt_out, out,
            CBL, CDI, CDM, 1,
            0, 0, 0, 0, 0, 0, 1);
    }
}
